// CrossAttention_54065048323025
// MI455X (gfx1250) — matrix operand order repaired and checked by reading, not compiled, not run on hardware
//
#include <hip/hip_runtime.h>

// ---------------------------------------------------------------------------
// CDNA5 (gfx1250) cross-attention forward, bf16 WMMA + flash-attention with
// async global->LDS double-buffered staging.
// Shapes: B=4, Sd=Se=2048, E=1024, H=16, D=64.
// ---------------------------------------------------------------------------

#define BATCH 4
#define SD    2048
#define SE    2048
#define EMB   1024
#define HEADS 16
#define HDIM  64

typedef __attribute__((ext_vector_type(16))) __bf16 v16bf;
typedef __attribute__((ext_vector_type(8)))  float  v8f;
typedef int v4i __attribute__((vector_size(16)));

__device__ __forceinline__ unsigned short f2bf(float x) {
  unsigned u = __float_as_uint(x);
  u += 0x7FFFu + ((u >> 16) & 1u);           // round-to-nearest-even
  return (unsigned short)(u >> 16);
}

// Load a 16-element bf16 fragment from two 16-byte chunks (global or LDS).
__device__ __forceinline__ v16bf ld_frag2(const unsigned short* p0,
                                          const unsigned short* p1) {
  union { v16bf v; uint4 q[2]; } u;
  u.q[0] = *reinterpret_cast<const uint4*>(p0);
  u.q[1] = *reinterpret_cast<const uint4*>(p1);
  return u.v;
}

__device__ __forceinline__ v8f wmma_bf16(v16bf a, v16bf b, v8f c) {
  return __builtin_amdgcn_wmma_f32_16x16x32_bf16(
      /*neg_a=*/false, a, /*neg_b=*/false, b,
      /*c_mod=*/(short)0, c, /*reuse_a=*/false, /*reuse_b=*/false);
}

// ---- CDNA5 async global->LDS copy (ASYNCcnt) with sync fallback -----------
#if __has_builtin(__builtin_amdgcn_global_load_async_to_lds_b128)
#define HAVE_ASYNC_LDS 1
#else
#define HAVE_ASYNC_LDS 0
#endif

__device__ __forceinline__ void async_b128(const unsigned short* g,
                                           unsigned short* l) {
#if HAVE_ASYNC_LDS
  __builtin_amdgcn_global_load_async_to_lds_b128(
      (__attribute__((address_space(1))) v4i*)g,
      (__attribute__((address_space(3))) v4i*)l, 0, 0);
#else
  *reinterpret_cast<uint4*>(l) = *reinterpret_cast<const uint4*>(g);
#endif
}

__device__ __forceinline__ void wait_async_all() {
#if HAVE_ASYNC_LDS
#if __has_builtin(__builtin_amdgcn_s_wait_asynccnt)
  __builtin_amdgcn_s_wait_asynccnt(0);
#endif
#endif
}

// ---------------------------------------------------------------------------
// fp32 -> bf16 elementwise convert (4 elems / thread)
// ---------------------------------------------------------------------------
__global__ __launch_bounds__(256) void cvt_f32_bf16(
    const float* __restrict__ in, unsigned short* __restrict__ out, int n) {
  int i = (blockIdx.x * blockDim.x + threadIdx.x) * 4;
  if (i + 3 < n) {
    float4 f = *reinterpret_cast<const float4*>(in + i);
    unsigned short __align__(8) t[4];
    t[0] = f2bf(f.x); t[1] = f2bf(f.y); t[2] = f2bf(f.z); t[3] = f2bf(f.w);
    *reinterpret_cast<uint2*>(out + i) = *reinterpret_cast<uint2*>(t);
  }
}

// ---------------------------------------------------------------------------
// 1024x1024 fp32 -> bf16 transpose (WT[n][k] = W[k][n]) via LDS tile
// ---------------------------------------------------------------------------
__global__ __launch_bounds__(256) void transpose_bf16(
    const float* __restrict__ W, unsigned short* __restrict__ WT) {
  __shared__ float tile[32][33];
  const int bx = blockIdx.x * 32, by = blockIdx.y * 32;
  const int tx = threadIdx.x & 31, ty = threadIdx.x >> 5;   // 32x8
#pragma unroll
  for (int i = 0; i < 32; i += 8)
    tile[ty + i][tx] = W[(by + ty + i) * EMB + bx + tx];
  __syncthreads();
#pragma unroll
  for (int i = 0; i < 32; i += 8)
    WT[(bx + ty + i) * EMB + by + tx] = f2bf(tile[tx][ty + i]);
}

// ---------------------------------------------------------------------------
// bf16 WMMA GEMM: C[M x 1024] = A[M x 1024] * B (B pre-transposed: BT[n][k]).
// One wave -> 16x64 C tile; the 64x32 BT k-slab is staged in LDS once per
// block (8 waves share it), async + double buffered.  All four B fragments
// are loaded into distinct registers before the WMMA batch so the matrix
// pipe issues back-to-back instead of stalling on DScnt per WMMA.
// blockDim = 256, grid = (N/64, M/128).
// mode 0: store bf16 row-major
// mode 1: store bf16 transposed per batch: C[(b*EMB + n)*SE + s]  (V path)
// mode 2: store f32 row-major + bias[n]   (output projection)
// ---------------------------------------------------------------------------
__global__ __launch_bounds__(256) void gemm_bf16_wmma(
    const unsigned short* __restrict__ A, const unsigned short* __restrict__ BT,
    void* __restrict__ Cp, const float* __restrict__ bias, int mode) {
  __shared__ __align__(16) unsigned short Blds[2][64 * 32];   // 8 KB

  const int tid = threadIdx.x;
  const int lane = tid & 31;
  const int wave = tid >> 5;
  const int ll = lane & 15, hi = lane >> 4;
  const int m0 = (blockIdx.y * 8 + wave) * 16;
  const int n0 = blockIdx.x * 64;

  // staging: thread t copies 16B chunk t of the 64x32 slab
  const int srow = tid >> 2;              // 0..63 (local n)
  const int scol = (tid & 3) * 8;         // 0,8,16,24 (k within slab)
  const unsigned short* gB = BT + (size_t)(n0 + srow) * EMB + scol;
  auto stageB = [&](int buf, int kk) {
    async_b128(gB + kk, &Blds[buf][srow * 32 + scol]);
  };

  const unsigned short* arow = A + (size_t)(m0 + ll) * EMB;

  v8f acc[4];
#pragma unroll
  for (int j = 0; j < 4; ++j)
#pragma unroll
    for (int r = 0; r < 8; ++r) acc[j][r] = 0.f;

  stageB(0, 0);
  int buf = 0;
  for (int kk = 0; kk < EMB; kk += 32, buf ^= 1) {
    wait_async_all();
    __syncthreads();
    if (kk + 32 < EMB) stageB(buf ^ 1, kk + 32);

    const unsigned short* pa = arow + kk + (hi ? 16 : 0);
    v16bf a = ld_frag2(pa, pa + 8);
    v16bf bf[4];
#pragma unroll
    for (int j = 0; j < 4; ++j) {
      const unsigned short* pb = &Blds[buf][(16 * j + ll) * 32 + (hi ? 16 : 0)];
      bf[j] = ld_frag2(pb, pb + 8);
    }
#pragma unroll
    for (int j = 0; j < 4; ++j) acc[j] = wmma_bf16(a, bf[j], acc[j]);
  }

  if (mode == 0) {
    unsigned short* C = (unsigned short*)Cp;
#pragma unroll
    for (int j = 0; j < 4; ++j) {
      const int n = n0 + 16 * j + ll;
#pragma unroll
      for (int r = 0; r < 8; ++r)
        C[(size_t)(m0 + r + 8 * hi) * EMB + n] = f2bf(acc[j][r]);
    }
  } else if (mode == 1) {
    unsigned short* C = (unsigned short*)Cp;
    const int t0 = m0 + 8 * hi;          // token index of r=0
    const int b = t0 >> 11;              // /2048
    const int s = t0 & 2047;
#pragma unroll
    for (int j = 0; j < 4; ++j) {
      const int n = n0 + 16 * j + ll;
      unsigned short __align__(16) t[8];
#pragma unroll
      for (int r = 0; r < 8; ++r) t[r] = f2bf(acc[j][r]);
      *reinterpret_cast<uint4*>(&C[((size_t)b * EMB + n) * SE + s]) =
          *reinterpret_cast<uint4*>(t);
    }
  } else {
    float* C = (float*)Cp;
#pragma unroll
    for (int j = 0; j < 4; ++j) {
      const int n = n0 + 16 * j + ll;
      const float bv = bias ? bias[n] : 0.f;
#pragma unroll
      for (int r = 0; r < 8; ++r)
        C[(size_t)(m0 + r + 8 * hi) * EMB + n] = acc[j][r] + bv;
    }
  }
}

// ---------------------------------------------------------------------------
// Flash attention: 4 waves/block, each wave one 16-row Q block of one (b,h);
// the 32-token K and Vt tiles are staged in LDS once per block (shared by all
// 4 waves), async + double buffered.
// Q,K: bf16 [B*S][EMB] row-major; Vt: bf16 [B][EMB][SE]; Ob: bf16 [B*Sd][EMB]
// blockDim = 128, grid = (Sd/64, B*H)
// ---------------------------------------------------------------------------
__global__ __launch_bounds__(128) void attn_flash_wmma(
    const unsigned short* __restrict__ Qb, const unsigned short* __restrict__ Kb,
    const unsigned short* __restrict__ Vt, unsigned short* __restrict__ Ob) {
  __shared__ __align__(16) unsigned short Klds[2][32 * 64];   // 8 KB
  __shared__ __align__(16) unsigned short Vlds[2][64 * 32];   // 8 KB
  __shared__ __align__(16) unsigned short Plds[4][16 * 32];   // 4 KB

  const int tid = threadIdx.x;
  const int lane = tid & 31;
  const int wave = tid >> 5;
  const int ll = lane & 15, hi = lane >> 4;
  const int b = blockIdx.y >> 4;
  const int h = blockIdx.y & 15;
  const int q0 = (blockIdx.x * 4 + wave) * 16;

  // ---- staging: 256 K-chunks (32 rows x 64 d) + 256 V-chunks (64 d x 32 s),
  //      2 of each per thread per tile.
  auto stageKV = [&](int buf, int s0) {
#pragma unroll
    for (int i = 0; i < 2; ++i) {
      const int c = tid + i * 128;
      const int kr = c >> 3, kc = (c & 7) * 8;
      async_b128(Kb + ((size_t)(b * SE + s0 + kr)) * EMB + h * HDIM + kc,
                 &Klds[buf][kr * 64 + kc]);
      const int vr = c >> 2, vc = (c & 3) * 8;
      async_b128(Vt + ((size_t)(b * EMB + h * HDIM + vr)) * SE + s0 + vc,
                 &Vlds[buf][vr * 32 + vc]);
    }
  };

  // ---- preload Q A-fragments (two 32-wide halves of d) --------------------
  v16bf aq[2];
#pragma unroll
  for (int dh = 0; dh < 2; ++dh) {
    const unsigned short* p = Qb + ((size_t)(b * SD + q0 + ll)) * EMB +
                              h * HDIM + dh * 32 + (hi ? 16 : 0);
    aq[dh] = ld_frag2(p, p + 8);
  }

  float mrow[8], lrow[8];
  v8f o[4];
#pragma unroll
  for (int r = 0; r < 8; ++r) { mrow[r] = -3.0e38f; lrow[r] = 0.f; }
#pragma unroll
  for (int dj = 0; dj < 4; ++dj)
#pragma unroll
    for (int r = 0; r < 8; ++r) o[dj][r] = 0.f;

  stageKV(0, 0);
  int buf = 0;
  for (int s0 = 0; s0 < SE; s0 += 32, buf ^= 1) {
    wait_async_all();
    __syncthreads();
    if (s0 + 32 < SE) stageKV(buf ^ 1, s0 + 32);

    // ---- S = (Q K^T) * 1/sqrt(D): two 16x16 score tiles from LDS ---------
    v16bf kf[4];
#pragma unroll
    for (int j = 0; j < 2; ++j)
#pragma unroll
      for (int dh = 0; dh < 2; ++dh) {
        const unsigned short* p =
            &Klds[buf][(16 * j + ll) * 64 + dh * 32 + (hi ? 16 : 0)];
        kf[j * 2 + dh] = ld_frag2(p, p + 8);
      }
    v8f s[2];
#pragma unroll
    for (int j = 0; j < 2; ++j) {
#pragma unroll
      for (int r = 0; r < 8; ++r) s[j][r] = 0.f;
#pragma unroll
      for (int dh = 0; dh < 2; ++dh)
        s[j] = wmma_bf16(aq[dh], kf[j * 2 + dh], s[j]);
#pragma unroll
      for (int r = 0; r < 8; ++r) s[j][r] *= 0.125f;   // 1/sqrt(64)
    }

    // ---- online softmax: row max / sum across the 16-lane groups ---------
#pragma unroll
    for (int r = 0; r < 8; ++r) {
      float rmax = fmaxf(s[0][r], s[1][r]);
#pragma unroll
      for (int m = 8; m >= 1; m >>= 1) rmax = fmaxf(rmax, __shfl_xor(rmax, m));
      const float mn = fmaxf(mrow[r], rmax);
      const float sc = __expf(mrow[r] - mn);
      const float p0 = __expf(s[0][r] - mn);
      const float p1 = __expf(s[1][r] - mn);
      s[0][r] = p0; s[1][r] = p1;
      float rs = p0 + p1;
#pragma unroll
      for (int m = 8; m >= 1; m >>= 1) rs += __shfl_xor(rs, m);
      lrow[r] = lrow[r] * sc + rs;
      mrow[r] = mn;
#pragma unroll
      for (int dj = 0; dj < 4; ++dj) o[dj][r] *= sc;
    }

    // ---- restage P: C-layout (f32) -> A-layout (bf16) via wave-private LDS
    //      (same-wave DS ops are in-order; no block barrier needed)
    unsigned short* pw = Plds[wave];
#pragma unroll
    for (int j = 0; j < 2; ++j)
#pragma unroll
      for (int r = 0; r < 8; ++r)
        pw[(r + 8 * hi) * 32 + 16 * j + ll] = f2bf(s[j][r]);
    const unsigned short* pr = &pw[ll * 32 + (hi ? 16 : 0)];
    v16bf ap = ld_frag2(pr, pr + 8);

    // ---- O += P @ V from LDS (Vt tile is [d][token]) ---------------------
    v16bf vf[4];
#pragma unroll
    for (int dj = 0; dj < 4; ++dj) {
      const unsigned short* p =
          &Vlds[buf][(dj * 16 + ll) * 32 + (hi ? 16 : 0)];
      vf[dj] = ld_frag2(p, p + 8);
    }
#pragma unroll
    for (int dj = 0; dj < 4; ++dj) o[dj] = wmma_bf16(ap, vf[dj], o[dj]);
  }

  // ---- normalize and store O (bf16, [B*Sd][EMB]) --------------------------
#pragma unroll
  for (int dj = 0; dj < 4; ++dj) {
    const int n = h * HDIM + dj * 16 + ll;
#pragma unroll
    for (int r = 0; r < 8; ++r) {
      const float v = o[dj][r] / lrow[r];
      Ob[((size_t)(b * SD + q0 + r + 8 * hi)) * EMB + n] = f2bf(v);
    }
  }
}

// ---------------------------------------------------------------------------
// Host-side launch
// ---------------------------------------------------------------------------
extern "C" void kernel_launch(void* const* d_in, const int* in_sizes, int n_in,
                              void* d_out, int out_size, void* d_ws, size_t ws_size,
                              hipStream_t stream) {
  const float* dec = (const float*)d_in[0];
  const float* enc = (const float*)d_in[1];
  const float* W_q = (const float*)d_in[2];
  const float* W_k = (const float*)d_in[3];
  const float* W_v = (const float*)d_in[4];
  const float* W_o = (const float*)d_in[5];
  const float* b_o = (const float*)d_in[6];
  float* out = (float*)d_out;

  char* ws = (char*)d_ws;
  const size_t ACT  = (size_t)BATCH * SD * EMB * sizeof(unsigned short); // 16 MiB
  const size_t WSZ  = (size_t)EMB * EMB * sizeof(unsigned short);        //  2 MiB
  unsigned short* decB = (unsigned short*)(ws);
  unsigned short* encB = (unsigned short*)(ws + ACT);
  unsigned short* WqT  = (unsigned short*)(ws + 2 * ACT);
  unsigned short* WkT  = (unsigned short*)(ws + 2 * ACT + WSZ);
  unsigned short* WvT  = (unsigned short*)(ws + 2 * ACT + 2 * WSZ);
  unsigned short* WoT  = (unsigned short*)(ws + 2 * ACT + 3 * WSZ);
  unsigned short* Qb   = (unsigned short*)(ws + 2 * ACT + 4 * WSZ);
  unsigned short* Kb   = (unsigned short*)(ws + 3 * ACT + 4 * WSZ);
  unsigned short* Vt   = (unsigned short*)(ws + 4 * ACT + 4 * WSZ);
  unsigned short* Ob   = (unsigned short*)(ws + 5 * ACT + 4 * WSZ);

  const int nAct = BATCH * SD * EMB;                   // 8388608
  // 1) activations fp32 -> bf16
  cvt_f32_bf16<<<nAct / (256 * 4), 256, 0, stream>>>(dec, decB, nAct);
  cvt_f32_bf16<<<nAct / (256 * 4), 256, 0, stream>>>(enc, encB, nAct);
  // 2) weights fp32 -> bf16 transposed
  dim3 tg(32, 32);
  transpose_bf16<<<tg, 256, 0, stream>>>(W_q, WqT);
  transpose_bf16<<<tg, 256, 0, stream>>>(W_k, WkT);
  transpose_bf16<<<tg, 256, 0, stream>>>(W_v, WvT);
  transpose_bf16<<<tg, 256, 0, stream>>>(W_o, WoT);
  // 3) projections (M = B*S = 8192, N = K = 1024)
  dim3 gg(EMB / 64, (BATCH * SD) / 128);               // (16, 64)
  gemm_bf16_wmma<<<gg, 256, 0, stream>>>(decB, WqT, Qb, nullptr, 0);
  gemm_bf16_wmma<<<gg, 256, 0, stream>>>(encB, WkT, Kb, nullptr, 0);
  gemm_bf16_wmma<<<gg, 256, 0, stream>>>(encB, WvT, Vt, nullptr, 1);
  // 4) fused flash attention
  dim3 ga(SD / 64, BATCH * HEADS);                     // (32, 64)
  attn_flash_wmma<<<ga, 128, 0, stream>>>(Qb, Kb, Vt, Ob);
  // 5) output projection + bias -> fp32
  gemm_bf16_wmma<<<gg, 256, 0, stream>>>(Ob, WoT, out, b_o, 2);
}
